// MutliHeadSelfAttention_26113401160415
// MI455X (gfx1250) — matrix operand order repaired and checked by reading, not compiled, not run on hardware
//
#include <hip/hip_runtime.h>

// ---------------------------------------------------------------------------
// MultiHeadSelfAttention, MI455X (gfx1250, wave32, WMMA bf16 16x16x32,
// async global->LDS double-buffered GEMM pipeline).
// B=2, S=2048, D=2048, H=16, HD=128.
//
// Reference's masking degenerates to:
//   attn[i] = ( sum_{j<i} v[j] + exp(s_ii) * v[i] ) / D_i,
//   D_i = i + sum_{r>=i} exp(s_ir),  s_ir = q_i . k_r / sqrt(D)
// ---------------------------------------------------------------------------

typedef __attribute__((ext_vector_type(16))) __bf16 v16bf;
typedef __attribute__((ext_vector_type(8)))  __bf16 v8bf;
typedef __attribute__((ext_vector_type(4)))  __bf16 v4bf;
typedef __attribute__((ext_vector_type(8)))  float  v8f;

constexpr int kB  = 2;
constexpr int kS  = 2048;
constexpr int kD  = 2048;
constexpr int kH  = 16;
constexpr int kHD = 128;
constexpr int kM  = kB * kS;          // 4096 total rows
constexpr int kLdsStride = 40;        // 32 payload + 8 pad elems: conflict-free b128 reads
constexpr float kScale = 0.02209708691207961f;  // 1/sqrt(2048)

// ---- WMMA wrapper (guarded so the host pass never sees the builtin) -------
__device__ inline v8f wmma_bf16(v16bf a, v16bf b, v8f c) {
#if defined(__HIP_DEVICE_COMPILE__)
  return __builtin_amdgcn_wmma_f32_16x16x32_bf16(
      /*neg_a=*/false, a, /*neg_b=*/false, b,
      /*c_mod=*/(short)0, c, /*reuse_a=*/false, /*reuse_b=*/false);
#else
  (void)a; (void)b;
  return c;
#endif
}

// ---- CDNA5 async global->LDS copy (ASYNCcnt-tracked), inline asm ----------
// Generic LDS pointer low 32 bits == group-segment byte offset (ISA 10.2).
__device__ inline unsigned lds_off(const void* p) {
  return (unsigned)(uintptr_t)p;
}
__device__ inline void async_copy_b128(unsigned lds, const __bf16* g) {
#if defined(__HIP_DEVICE_COMPILE__)
  asm volatile("global_load_async_to_lds_b128 %0, %1, off"
               :: "v"(lds), "v"(g) : "memory");
#else
  (void)lds; (void)g;
#endif
}
__device__ inline void wait_asynccnt0() {
#if defined(__HIP_DEVICE_COMPILE__)
  asm volatile("s_wait_asynccnt 0x0" ::: "memory");
#endif
}
__device__ inline void wait_asynccnt4() {
#if defined(__HIP_DEVICE_COMPILE__)
  asm volatile("s_wait_asynccnt 0x4" ::: "memory");
#endif
}

// ---- Fragment loaders per ISA 7.12.2 lane layouts (wave32) ----------------
// A (16x32 bf16): lane = half*16+m holds row m; K-base = half*8;
// elements 0..7 -> K=kb..kb+7, elements 8..15 -> K=kb+16..kb+23.
__device__ inline v16bf frag_a_from(const __bf16* __restrict__ p0, int ld) {
  const int lane = threadIdx.x & 31;
  const int m  = lane & 15;
  const int kb = (lane >> 4) << 3;            // 0 or 8
  const __bf16* p = p0 + (size_t)m * ld + kb;
  v8bf lo = *(const v8bf*)(p);
  v8bf hi = *(const v8bf*)(p + 16);
  v16bf a;
#pragma unroll
  for (int e = 0; e < 8; ++e) { a[e] = lo[e]; a[e + 8] = hi[e]; }
  return a;
}

// B (32x16 bf16): lane = half*16+n holds column n; K-base = half*8;
// elements 0..7 -> K=kb..kb+7, elements 8..15 -> K=kb+16..kb+23.
// Column n of B == row n of the row-major [N,K] source (y = x * W^T).
__device__ inline v16bf frag_b_from(const __bf16* __restrict__ p0, int ld) {
  const int lane = threadIdx.x & 31;
  const int n  = lane & 15;
  const int kb = (lane >> 4) << 3;            // 0 or 8
  const __bf16* p = p0 + (size_t)n * ld + kb;
  v8bf lo = *(const v8bf*)(p);
  v8bf hi = *(const v8bf*)(p + 16);
  v16bf b;
#pragma unroll
  for (int e = 0; e < 8; ++e) { b[e] = lo[e]; b[e + 8] = hi[e]; }
  return b;
}

// ---- f32 -> bf16 cast (4 elems/thread) ------------------------------------
__global__ __launch_bounds__(256) void cast_kernel(
    const float* __restrict__ in, __bf16* __restrict__ out, int n) {
  int idx = (blockIdx.x * 256 + threadIdx.x) * 4;
  if (idx + 3 < n) {
    float4 f = *(const float4*)(in + idx);
    v4bf o;
    o[0] = (__bf16)f.x; o[1] = (__bf16)f.y;
    o[2] = (__bf16)f.z; o[3] = (__bf16)f.w;
    *(v4bf*)(out + idx) = o;
  }
}

// ---- C[m,n] = sum_k A[m,k]*W[n,k] + bias[n] -------------------------------
// Block 128x128, 8 waves (4 m-slots x 2 n-slots), wave tile 32x64 = 2x4
// WMMA accumulators. K-step 32. Tiles staged into LDS with async copies,
// double-buffered against the WMMA pipeline.
template <typename OutT>
__global__ __launch_bounds__(256) void gemm_bf16_wmma(
    const __bf16* __restrict__ A, const __bf16* __restrict__ W,
    const float* __restrict__ bias, OutT* __restrict__ C,
    int M, int N, int K) {
  __shared__ __bf16 sA[2][128 * kLdsStride];
  __shared__ __bf16 sB[2][128 * kLdsStride];

  const int tid  = threadIdx.x;
  const int wave = tid >> 5;
  const int lane = tid & 31;
  const int wm   = wave & 3;
  const int wn   = wave >> 2;
  const int bm   = blockIdx.y * 128;
  const int bn   = blockIdx.x * 128;

  // Each thread copies two 16B chunks per 128x32 tile (512 chunks, 256 thr).
  const int r0c = tid >> 2,          s0c = tid & 3;          // chunk tid
  const int r1c = (tid + 256) >> 2,  s1c = (tid + 256) & 3;  // chunk tid+256

  auto issue = [&](int buf, int k0) {
    async_copy_b128(lds_off(&sA[buf][r0c * kLdsStride + s0c * 8]),
                    A + (size_t)(bm + r0c) * K + k0 + s0c * 8);
    async_copy_b128(lds_off(&sA[buf][r1c * kLdsStride + s1c * 8]),
                    A + (size_t)(bm + r1c) * K + k0 + s1c * 8);
    async_copy_b128(lds_off(&sB[buf][r0c * kLdsStride + s0c * 8]),
                    W + (size_t)(bn + r0c) * K + k0 + s0c * 8);
    async_copy_b128(lds_off(&sB[buf][r1c * kLdsStride + s1c * 8]),
                    W + (size_t)(bn + r1c) * K + k0 + s1c * 8);
  };

  v8f acc[2][4];
#pragma unroll
  for (int i = 0; i < 2; ++i)
#pragma unroll
    for (int j = 0; j < 4; ++j) acc[i][j] = {};

  const int niter = K >> 5;
  issue(0, 0);
  for (int it = 0; it < niter; ++it) {
    const int cur = it & 1;
    if (it + 1 < niter) {
      issue(cur ^ 1, (it + 1) << 5);  // prefetch next K-slab
      wait_asynccnt4();               // in-order: the 4 older (cur) are done
    } else {
      wait_asynccnt0();
    }
    __syncthreads();                  // all waves' cur writes visible

    v16bf af[2], bfr[4];
#pragma unroll
    for (int i = 0; i < 2; ++i)
      af[i] = frag_a_from(&sA[cur][(wm * 32 + i * 16) * kLdsStride], kLdsStride);
#pragma unroll
    for (int j = 0; j < 4; ++j)
      bfr[j] = frag_b_from(&sB[cur][(wn * 64 + j * 16) * kLdsStride], kLdsStride);
#pragma unroll
    for (int i = 0; i < 2; ++i)
#pragma unroll
      for (int j = 0; j < 4; ++j)
        acc[i][j] = wmma_bf16(af[i], bfr[j], acc[i][j]);

    __syncthreads();                  // readers done before cur is re-filled
  }

  const int nlane = lane & 15;
  const int mhi   = (lane >> 4) << 3;         // 0 or 8
#pragma unroll
  for (int j = 0; j < 4; ++j) {
    const int col = bn + wn * 64 + j * 16 + nlane;
    const float bv = bias[col];
#pragma unroll
    for (int i = 0; i < 2; ++i) {
#pragma unroll
      for (int r = 0; r < 8; ++r) {
        const int row = bm + wm * 32 + i * 16 + mhi + r;
        C[(size_t)row * N + col] = (OutT)(acc[i][j][r] + bv);
      }
    }
  }
}

// ---- Attention denominators + diagonal ------------------------------------
// grid (S/128, B*H); each wave owns 16 query rows, streams 16-key tiles
// starting at the diagonal. Scores via 4 WMMAs (K=128 = 4 x 32).
__global__ __launch_bounds__(256) void attn_denom_kernel(
    const __bf16* __restrict__ Qb, const __bf16* __restrict__ Kb,
    float* __restrict__ denom, float* __restrict__ diagexp) {
  const int bh   = blockIdx.y;                // b*H + h
  const int b    = bh >> 4;
  const int h    = bh & 15;
  const int wave = threadIdx.x >> 5;
  const int lane = threadIdx.x & 31;
  const int i0   = blockIdx.x * 128 + wave * 16;
  const size_t rowbase = (size_t)b * kS;
  const int colbase    = h * kHD;

  // 16 query rows x 128 dims = 4 reusable A fragments.
  v16bf qf[4];
#pragma unroll
  for (int kc = 0; kc < 4; ++kc)
    qf[kc] = frag_a_from(Qb + (rowbase + i0) * kD + colbase + kc * 32, kD);

  float rowsum[8], dg[8];
#pragma unroll
  for (int r = 0; r < 8; ++r) { rowsum[r] = 0.f; dg[r] = 0.f; }

  const int nlane = lane & 15;
  const int mhi   = (lane >> 4) << 3;

  for (int r0 = i0; r0 < kS; r0 += 16) {      // only r >= i tiles matter
    v8f acc = {};
#pragma unroll
    for (int kc = 0; kc < 4; ++kc) {
      v16bf kf = frag_b_from(Kb + (rowbase + r0) * kD + colbase + kc * 32, kD);
      acc = wmma_bf16(qf[kc], kf, acc);
    }
    const int rkey = r0 + nlane;
    if (r0 == i0) {                           // diagonal (partial) tile
#pragma unroll
      for (int r = 0; r < 8; ++r) {
        const int i = i0 + mhi + r;
        if (rkey >= i) {
          const float ex = __expf(acc[r] * kScale);
          rowsum[r] += ex;
          if (rkey == i) dg[r] = ex;
        }
      }
    } else {                                  // fully unmasked tile
#pragma unroll
      for (int r = 0; r < 8; ++r) rowsum[r] += __expf(acc[r] * kScale);
    }
  }

  // Reduce across the 16 key-lanes of each wave half.
#pragma unroll
  for (int r = 0; r < 8; ++r) {
#pragma unroll
    for (int off = 8; off >= 1; off >>= 1) {
      rowsum[r] += __shfl_xor(rowsum[r], off, 16);
      dg[r]     += __shfl_xor(dg[r], off, 16);
    }
  }
  if (nlane == 0) {
#pragma unroll
    for (int r = 0; r < 8; ++r) {
      const int i = i0 + mhi + r;
      denom[(size_t)bh * kS + i]   = (float)i + rowsum[r];
      diagexp[(size_t)bh * kS + i] = dg[r];
    }
  }
}

// ---- Prefix-sum of V over sequence (blocked scan, 128-row blocks) ---------
__global__ __launch_bounds__(256) void vpartial_kernel(
    const float* __restrict__ V, float* __restrict__ bsum) {
  const int c  = blockIdx.x * 256 + threadIdx.x;
  const int sb = blockIdx.y;
  const int b  = blockIdx.z;
  const float* p = V + ((size_t)b * kS + sb * 128) * kD + c;
  float s = 0.f;
  for (int t = 0; t < 128; ++t) s += p[(size_t)t * kD];
  bsum[((size_t)b * 16 + sb) * kD + c] = s;
}

__global__ __launch_bounds__(256) void vscan_kernel(
    const float* __restrict__ bsum, float* __restrict__ boff) {
  const int c = blockIdx.x * 256 + threadIdx.x;
  const int b = blockIdx.y;
  float run = 0.f;
  for (int sb = 0; sb < 16; ++sb) {
    const size_t idx = ((size_t)b * 16 + sb) * kD + c;
    const float t = bsum[idx];
    boff[idx] = run;
    run += t;
  }
}

// ---- Combine: attn_bf16 = (prefixV + diagexp*V) / denom -------------------
__global__ __launch_bounds__(256) void combine_kernel(
    const float* __restrict__ V, const float* __restrict__ boff,
    const float* __restrict__ denom, const float* __restrict__ diagexp,
    __bf16* __restrict__ attnb) {
  const int c  = blockIdx.x * 256 + threadIdx.x;
  const int sb = blockIdx.y;
  const int b  = blockIdx.z;
  const int h  = c >> 7;                       // c / HD
  float run = boff[((size_t)b * 16 + sb) * kD + c];
  const size_t dbase = ((size_t)b * kH + h) * kS;
  for (int t = 0; t < 128; ++t) {
    const int i = sb * 128 + t;
    const size_t vidx = ((size_t)b * kS + i) * kD + c;
    const float v  = V[vidx];
    const float de = denom[dbase + i];
    const float dx = diagexp[dbase + i];
    attnb[vidx] = (__bf16)((run + dx * v) / de);
    run += v;
  }
}

// ---------------------------------------------------------------------------
extern "C" void kernel_launch(void* const* d_in, const int* in_sizes, int n_in,
                              void* d_out, int out_size, void* d_ws, size_t ws_size,
                              hipStream_t stream) {
  (void)in_sizes; (void)n_in; (void)out_size; (void)ws_size;
  const float* x    = (const float*)d_in[0];
  const float* wq_w = (const float*)d_in[1];
  const float* wq_b = (const float*)d_in[2];
  const float* wk_w = (const float*)d_in[3];
  const float* wk_b = (const float*)d_in[4];
  const float* wv_w = (const float*)d_in[5];
  const float* wv_b = (const float*)d_in[6];
  const float* wo_w = (const float*)d_in[7];
  const float* wo_b = (const float*)d_in[8];
  float* out = (float*)d_out;

  char* ws = (char*)d_ws;
  const size_t MB = 1ull << 20;
  __bf16* xb    = (__bf16*)(ws);
  __bf16* wqb   = (__bf16*)(ws + 16 * MB);
  __bf16* wkb   = (__bf16*)(ws + 24 * MB);
  __bf16* wvb   = (__bf16*)(ws + 32 * MB);
  __bf16* wob   = (__bf16*)(ws + 40 * MB);
  __bf16* Qb    = (__bf16*)(ws + 48 * MB);
  __bf16* Kb    = (__bf16*)(ws + 64 * MB);
  float*  Vf    = (float*) (ws + 80 * MB);
  float*  denom = (float*) (ws + 112 * MB);
  float*  dge   = (float*) (ws + 112 * MB + 256 * 1024);
  float*  bsum  = (float*) (ws + 112 * MB + 512 * 1024);
  float*  boff  = (float*) (ws + 112 * MB + 768 * 1024);
  __bf16* attnb = (__bf16*)(ws + 113 * MB);

  // Downconvert activations and weights to bf16.
  cast_kernel<<<(kM * kD) / 1024, 256, 0, stream>>>(x,    xb,  kM * kD);
  cast_kernel<<<(kD * kD) / 1024, 256, 0, stream>>>(wq_w, wqb, kD * kD);
  cast_kernel<<<(kD * kD) / 1024, 256, 0, stream>>>(wk_w, wkb, kD * kD);
  cast_kernel<<<(kD * kD) / 1024, 256, 0, stream>>>(wv_w, wvb, kD * kD);
  cast_kernel<<<(kD * kD) / 1024, 256, 0, stream>>>(wo_w, wob, kD * kD);

  const dim3 gg(kD / 128, kM / 128);           // (16, 32)
  // Q,K only ever feed WMMA scores -> store bf16 directly. V stays f32.
  gemm_bf16_wmma<__bf16><<<gg, 256, 0, stream>>>(xb, wqb, wq_b, Qb, kM, kD, kD);
  gemm_bf16_wmma<__bf16><<<gg, 256, 0, stream>>>(xb, wkb, wk_b, Kb, kM, kD, kD);
  gemm_bf16_wmma<float ><<<gg, 256, 0, stream>>>(xb, wvb, wv_b, Vf, kM, kD, kD);

  attn_denom_kernel<<<dim3(kS / 128, kB * kH), 256, 0, stream>>>(Qb, Kb, denom, dge);

  vpartial_kernel<<<dim3(kD / 256, 16, kB), 256, 0, stream>>>(Vf, bsum);
  vscan_kernel  <<<dim3(kD / 256, kB),      256, 0, stream>>>(bsum, boff);
  combine_kernel<<<dim3(kD / 256, 16, kB),  256, 0, stream>>>(Vf, boff, denom, dge, attnb);

  gemm_bf16_wmma<float><<<gg, 256, 0, stream>>>(attnb, wob, wo_b, out, kM, kD, kD);
}
